// OmnivoreSwinLayer_74723841016140
// MI455X (gfx1250) — matrix operand order repaired and checked by reading, not compiled, not run on hardware
//
#include <hip/hip_runtime.h>
#include <hip/hip_bf16.h>

typedef __attribute__((ext_vector_type(16))) _Float16 v16h;
typedef __attribute__((ext_vector_type(8)))  _Float16 v8h;
typedef __attribute__((ext_vector_type(8)))  float    v8f;

#define DIMC   384
#define NHEADS 12
#define HD     32
#define SWIN   98          // tokens per 3D window
#define NWIN   1024        // total windows (B * 512)
#define NTOK   100352      // B*D*H*W == NWIN*SWIN
#define SQ     112         // padded query rows (7*16)
#define SK     128         // padded key rows (4*32)

// ---------------------------------------------------------------------------
// WMMA helpers (gfx1250 wave32, 16x16x32 f16 -> f32)
// ---------------------------------------------------------------------------
__device__ __forceinline__ v8f wmma16x16x32(v16h a, v16h b, v8f c) {
  return __builtin_amdgcn_wmma_f32_16x16x32_f16(false, a, false, b, (short)0, c,
                                                false, false);
}

// A fragment: 16x32 f16. Lane r=lane&15 -> row M; halves j<8 : K=hi*8+j,
// halves j>=8 : K=16+hi*8+(j-8), hi=lane>>4.  (ISA 7.12.2)
__device__ __forceinline__ v16h frag_a(const _Float16* base, int ld, int lane) {
  const int r = lane & 15, hi = lane >> 4;
  const _Float16* p = base + r * ld + hi * 8;
  v8h lo = *(const v8h*)p;
  v8h h8 = *(const v8h*)(p + 16);
  v16h out;
#pragma unroll
  for (int i = 0; i < 8; ++i) { out[i] = lo[i]; out[8 + i] = h8[i]; }
  return out;
}

// B fragment: 32x16 f16, stored [n][k] (n-major, stride ld).
// Lane n=lane&15 -> col N; halves j<8 : K=hi*8+j, j>=8 : K=16+hi*8+(j-8).
__device__ __forceinline__ v16h frag_b(const _Float16* base, int ld, int lane) {
  const int n = lane & 15, hi = lane >> 4;
  const _Float16* p = base + n * ld + hi * 8;
  v8h lo = *(const v8h*)p;
  v8h h8 = *(const v8h*)(p + 16);
  v16h out;
#pragma unroll
  for (int i = 0; i < 8; ++i) { out[i] = lo[i]; out[8 + i] = h8[i]; }
  return out;
}

// Map windowed row (w*98+s) -> flat token index in (B,D,H,W) after
// window-reverse + reverse cyclic shift (also the LN1 gather map).
__device__ __forceinline__ size_t map_token(int row) {
  int w = row / SWIN, s = row - w * SWIN;
  int b = w >> 9, rem = w & 511;
  int zd = rem >> 6, zh = (rem >> 3) & 7, zw = rem & 7;
  int ld = s / 49, r49 = s - ld * 49, lh = r49 / 7, lw = r49 - lh * 7;
  int d0 = (zd * 2 + ld + 1) & 15;
  int h0 = zh * 7 + lh + 3; if (h0 >= 56) h0 -= 56;
  int w0 = zw * 7 + lw + 3; if (w0 >= 56) w0 -= 56;
  return (((size_t)b * 16 + d0) * 56 + h0) * 56 + w0;
}

// ---------------------------------------------------------------------------
// Small kernels
// ---------------------------------------------------------------------------
__global__ __launch_bounds__(256) void cvt_f16(const float* __restrict__ src,
                                               _Float16* __restrict__ dst, int n) {
  int i = blockIdx.x * 256 + threadIdx.x;
  if (i < n) dst[i] = (_Float16)src[i];
}

// Expand rel-pos-bias table -> per-head dense PADDED [12][112][128] matrices.
// Pad entries get -1e30 so the attention epilogue needs no bounds checks and
// padded keys vanish in softmax.
__global__ __launch_bounds__(256) void rpb_expand(const float* __restrict__ tab,
                                                  float* __restrict__ mat) {
  int i = blockIdx.x * 256 + threadIdx.x;           // 12*112*128 = 172032
  if (i >= NHEADS * SQ * SK) return;
  int head = i / (SQ * SK), rc = i - head * (SQ * SK);
  int r = rc >> 7, c = rc & 127;
  float v = -1e30f;
  if (r < SWIN && c < SWIN) {
    int qd = r / 49, qr = r - qd * 49, qh = qr / 7, qw = qr - qh * 7;
    int kd = c / 49, kr = c - kd * 49, kh = kr / 7, kw = kr - kh * 7;
    int ridx = (qd - kd + 1) * 169 + (qh - kh + 6) * 13 + (qw - kw + 6);
    v = tab[ridx * NHEADS + head];
  }
  mat[i] = v;
}

// Expand window mask (512,98,98) -> padded [512][112][128] with 0 in pads.
__global__ __launch_bounds__(256) void mask_expand(const float* __restrict__ mask,
                                                   float* __restrict__ mp) {
  int i = blockIdx.x * 256 + threadIdx.x;           // 512*112*128 = 7340032
  if (i >= 512 * SQ * SK) return;
  int w = i / (SQ * SK), rc = i - w * (SQ * SK);
  int r = rc >> 7, c = rc & 127;
  mp[i] = (r < SWIN && c < SWIN) ? mask[(size_t)w * SWIN * SWIN + r * SWIN + c] : 0.f;
}

// LN over C=384, one wave per token. mode 0: gather shifted window token from
// hidden_state -> xw[row].  mode 1: read x (token order) -> xh[row].
__global__ __launch_bounds__(256) void ln_kernel(const float* __restrict__ x,
                                                 const float* __restrict__ g,
                                                 const float* __restrict__ b,
                                                 _Float16* __restrict__ out,
                                                 int mode) {
  int wid = (blockIdx.x * 256 + threadIdx.x) >> 5;   // token / window-row id
  int lane = threadIdx.x & 31;
  size_t src_row = mode ? (size_t)wid : map_token(wid);
  const float* src = x + src_row * DIMC;
  float s = 0.f, ss = 0.f;
#pragma unroll
  for (int i = lane; i < DIMC; i += 32) { float v = src[i]; s += v; ss += v * v; }
#pragma unroll
  for (int o = 16; o; o >>= 1) { s += __shfl_xor(s, o, 32); ss += __shfl_xor(ss, o, 32); }
  float mean = s * (1.f / DIMC);
  float inv  = rsqrtf(ss * (1.f / DIMC) - mean * mean + 1e-5f);
  _Float16* dst = out + (size_t)wid * DIMC;
#pragma unroll
  for (int i = lane; i < DIMC; i += 32)
    dst[i] = (_Float16)((src[i] - mean) * inv * g[i] + b[i]);
}

// ---------------------------------------------------------------------------
// Generic WMMA GEMM: C[M,N] = A[M,K] (f16) x W[N,K]^T (f16), fused epilogue.
// Block = 256 threads (8 waves), tile 128(M) x 64(N), K step 32.
// ---------------------------------------------------------------------------
template <class Epi>
__global__ __launch_bounds__(256) void gemm128x64(const _Float16* __restrict__ A,
                                                  const _Float16* __restrict__ W,
                                                  int K, Epi epi) {
  __shared__ _Float16 As[128 * 40];
  __shared__ _Float16 Bs[64 * 40];
  const int tid = threadIdx.x;
  const int wave = tid >> 5, lane = tid & 31;
  const int mBase = blockIdx.y * 128;
  const int nBase = blockIdx.x * 64;
  v8f acc[4] = {};
  for (int k0 = 0; k0 < K; k0 += 32) {
    __syncthreads();
    {   // A tile: 128x32, 256 chunks of 16 halves
      int row = tid >> 1, off = (tid & 1) * 16;
      const _Float16* gp = A + (size_t)(mBase + row) * K + k0 + off;
      *(v8h*)(As + row * 40 + off)     = *(const v8h*)gp;
      *(v8h*)(As + row * 40 + off + 8) = *(const v8h*)(gp + 8);
    }
    if (tid < 128) {  // B tile: 64x32 from W[n][k]
      int row = tid >> 1, off = (tid & 1) * 16;
      const _Float16* gp = W + (size_t)(nBase + row) * K + k0 + off;
      *(v8h*)(Bs + row * 40 + off)     = *(const v8h*)gp;
      *(v8h*)(Bs + row * 40 + off + 8) = *(const v8h*)(gp + 8);
    }
    __syncthreads();
    v16h a = frag_a(As + wave * 16 * 40, 40, lane);
#pragma unroll
    for (int nt = 0; nt < 4; ++nt) {
      v16h bf = frag_b(Bs + nt * 16 * 40, 40, lane);
      acc[nt] = wmma16x16x32(a, bf, acc[nt]);
    }
  }
  const int hi = lane >> 4, c0 = lane & 15;
#pragma unroll
  for (int i = 0; i < 8; ++i)
#pragma unroll
    for (int nt = 0; nt < 4; ++nt)
      epi(mBase + wave * 16 + i + 8 * hi, nBase + nt * 16 + c0, acc[nt][i]);
}

// ------------------------------ epilogues ----------------------------------
struct EpiQKV {
  const float* bias; _Float16 *q, *k, *v; float scale;
  __device__ void operator()(int row, int col, float val) const {
    val += bias[col];
    int part = col / DIMC, hc = col - part * DIMC;
    int head = hc >> 5, dim = hc & 31;
    int w = row / SWIN, s = row - w * SWIN;
    size_t idx = (((size_t)w * NHEADS + head) * SWIN + s) * HD + dim;
    if (part == 0)      q[idx] = (_Float16)(val * scale);
    else if (part == 1) k[idx] = (_Float16)val;
    else                v[idx] = (_Float16)val;
  }
};

struct EpiProj {   // proj bias + window-reverse + roll + residual -> x (f32)
  const float* bias; const float* shortcut; float* out;
  __device__ void operator()(int row, int col, float val) const {
    size_t t = map_token(row);
    out[t * DIMC + col] = shortcut[t * DIMC + col] + val + bias[col];
  }
};

struct EpiGelu {   // mlp1 bias + exact GELU -> h (f16)
  const float* bias; _Float16* h;
  __device__ void operator()(int row, int col, float val) const {
    val += bias[col];
    val = 0.5f * val * (1.f + erff(val * 0.70710678118654752f));
    h[(size_t)row * 1536 + col] = (_Float16)val;
  }
};

struct EpiMlp2 {   // mlp2 bias + final residual accumulate into d_out
  const float* bias; float* out;
  __device__ void operator()(int row, int col, float val) const {
    out[(size_t)row * DIMC + col] += val + bias[col];
  }
};

// ---------------------------------------------------------------------------
// Windowed attention: one block per (window, head). Q:112x32, K/V padded 128.
// ---------------------------------------------------------------------------
#define ATTN_SMEM (SQ * SK * 4 + SQ * SK * 2 + SQ * HD * 2 + SK * HD * 2 + HD * SK * 2)

__global__ __launch_bounds__(256) void attn_kernel(const _Float16* __restrict__ qb,
                                                   const _Float16* __restrict__ kb,
                                                   const _Float16* __restrict__ vb,
                                                   const float* __restrict__ rpbm,
                                                   const float* __restrict__ maskp,
                                                   _Float16* __restrict__ out) {
  extern __shared__ char smem[];
  float*    Ss = (float*)smem;                                    // 112x128 f32
  _Float16* Ps = (_Float16*)(smem + SQ * SK * 4);                 // 112x128 f16
  _Float16* Qs = (_Float16*)(smem + SQ * SK * 6);                 // 112x32
  _Float16* Ks = (_Float16*)(smem + SQ * SK * 6 + SQ * HD * 2);   // 128x32
  _Float16* Vt = (_Float16*)(smem + SQ * SK * 6 + SQ * HD * 2 + SK * HD * 2); // 32x128

  const int wh = blockIdx.x;            // window*12 + head
  const int win = wh / NHEADS, head = wh - win * NHEADS;
  const int tid = threadIdx.x, wave = tid >> 5, lane = tid & 31;
  const int hi = lane >> 4, c0 = lane & 15;
  const _Float16* qsrc = qb + (size_t)wh * SWIN * HD;
  const _Float16* ksrc = kb + (size_t)wh * SWIN * HD;
  const _Float16* vsrc = vb + (size_t)wh * SWIN * HD;

  for (int i = tid; i < SQ * HD; i += 256) {
    int s = i >> 5, d = i & 31;
    Qs[i] = (s < SWIN) ? qsrc[s * HD + d] : (_Float16)0.f;
  }
  for (int i = tid; i < SK * HD; i += 256) {
    int s = i >> 5, d = i & 31;
    _Float16 kv = (s < SWIN) ? ksrc[s * HD + d] : (_Float16)0.f;
    _Float16 vv = (s < SWIN) ? vsrc[s * HD + d] : (_Float16)0.f;
    Ks[i] = kv;              // [key][dim] == B-layout [n][k] for Q*K^T
    Vt[d * SK + s] = vv;     // [dim][key] == B-layout [n][k] for P*V
  }
  __syncthreads();

  // scores = Q*K^T + rpb + mask (both pre-padded to 112x128, branchless)
  const float* maskw = maskp + (size_t)(win & 511) * SQ * SK;
  const float* rh    = rpbm + (size_t)head * SQ * SK;
  for (int t = wave; t < 56; t += 8) {
    int mt = t >> 3, nt = t & 7;
    v16h a = frag_a(Qs + mt * 16 * HD, HD, lane);
    v16h bf = frag_b(Ks + nt * 16 * HD, HD, lane);
    v8f acc = {};
    acc = wmma16x16x32(a, bf, acc);
#pragma unroll
    for (int i = 0; i < 8; ++i) {
      int rc = (mt * 16 + i + 8 * hi) * SK + nt * 16 + c0;
      Ss[rc] = acc[i] + rh[rc] + maskw[rc];
    }
  }
  __syncthreads();

  // softmax: wave per row (112 rows / 8 waves), lanes over 128 cols
  for (int r = wave; r < SQ; r += 8) {
    float* row = Ss + r * SK;
    float e0 = row[lane], e1 = row[lane + 32], e2 = row[lane + 64], e3 = row[lane + 96];
    float m = fmaxf(fmaxf(e0, e1), fmaxf(e2, e3));
#pragma unroll
    for (int o = 16; o; o >>= 1) m = fmaxf(m, __shfl_xor(m, o, 32));
    e0 = __expf(e0 - m); e1 = __expf(e1 - m); e2 = __expf(e2 - m); e3 = __expf(e3 - m);
    float sum = e0 + e1 + e2 + e3;
#pragma unroll
    for (int o = 16; o; o >>= 1) sum += __shfl_xor(sum, o, 32);
    float inv = 1.f / sum;
    _Float16* pr = Ps + r * SK;
    pr[lane]      = (_Float16)(e0 * inv);
    pr[lane + 32] = (_Float16)(e1 * inv);
    pr[lane + 64] = (_Float16)(e2 * inv);
    pr[lane + 96] = (_Float16)(e3 * inv);
  }
  __syncthreads();

  // out = P * V : 7x2 tiles, K=128 -> 4 WMMA steps
  for (int t = wave; t < 14; t += 8) {
    int mt = t >> 1, nt = t & 1;
    v8f acc = {};
#pragma unroll
    for (int kk = 0; kk < 4; ++kk) {
      v16h a = frag_a(Ps + mt * 16 * SK + kk * 32, SK, lane);
      v16h bf = frag_b(Vt + nt * 16 * SK + kk * 32, SK, lane);
      acc = wmma16x16x32(a, bf, acc);
    }
#pragma unroll
    for (int i = 0; i < 8; ++i) {
      int r = mt * 16 + i + 8 * hi;
      if (r < SWIN)
        out[((size_t)win * SWIN + r) * DIMC + head * HD + nt * 16 + c0] = (_Float16)acc[i];
    }
  }
}

// ---------------------------------------------------------------------------
// Launch: ws layout (halves)
// ---------------------------------------------------------------------------
static constexpr size_t OFF_WQKV  = 0;                      // 1152*384
static constexpr size_t OFF_WPROJ = 442368;                 // 384*384
static constexpr size_t OFF_W1    = 589824;                 // 1536*384
static constexpr size_t OFF_W2    = 1179648;                // 384*1536
static constexpr size_t OFF_RPB   = 1769472;                // 12*112*128 f32 = 344064 halves
static constexpr size_t OFF_MASK  = 2113536;                // 512*112*128 f32 = 14680064 halves
static constexpr size_t OFF_XW    = 16793600;               // 100352*384 (xw / attn_out / xh)
static constexpr size_t OFF_QKV   = 55328768;               // q,k,v then reused as mlp hidden
static constexpr size_t SZ_WT     = 38535168;               // 100352*384

extern "C" void kernel_launch(void* const* d_in, const int* in_sizes, int n_in,
                              void* d_out, int out_size, void* d_ws, size_t ws_size,
                              hipStream_t stream) {
  const float* hidden = (const float*)d_in[0];
  const float* mask   = (const float*)d_in[1];
  const float* n1w    = (const float*)d_in[2];
  const float* n1b    = (const float*)d_in[3];
  const float* qkv_w  = (const float*)d_in[4];
  const float* qkv_b  = (const float*)d_in[5];
  const float* rpb    = (const float*)d_in[6];
  const float* proj_w = (const float*)d_in[7];
  const float* proj_b = (const float*)d_in[8];
  const float* n2w    = (const float*)d_in[9];
  const float* n2b    = (const float*)d_in[10];
  const float* w1     = (const float*)d_in[11];
  const float* b1     = (const float*)d_in[12];
  const float* w2     = (const float*)d_in[13];
  const float* b2     = (const float*)d_in[14];
  float* out = (float*)d_out;
  _Float16* ws = (_Float16*)d_ws;

  _Float16* wqkv  = ws + OFF_WQKV;
  _Float16* wproj = ws + OFF_WPROJ;
  _Float16* wm1   = ws + OFF_W1;
  _Float16* wm2   = ws + OFF_W2;
  float*    rpbm  = (float*)(ws + OFF_RPB);
  float*    maskp = (float*)(ws + OFF_MASK);
  _Float16* xw    = ws + OFF_XW;     // LN1 windowed input (f16)
  _Float16* aout  = ws + OFF_XW;     // reuse: attention output (windowed)
  _Float16* xh    = ws + OFF_XW;     // reuse: LN2 output (token order)
  _Float16* qbuf  = ws + OFF_QKV;
  _Float16* kbuf  = qbuf + SZ_WT;
  _Float16* vbuf  = kbuf + SZ_WT;
  _Float16* hbuf  = ws + OFF_QKV;    // reuse q/k/v region: 100352*1536 halves

  // 1) weights -> f16; rel-pos bias & mask -> dense padded matrices
  cvt_f16<<<(442368 + 255) / 256, 256, 0, stream>>>(qkv_w,  wqkv,  442368);
  cvt_f16<<<(147456 + 255) / 256, 256, 0, stream>>>(proj_w, wproj, 147456);
  cvt_f16<<<(589824 + 255) / 256, 256, 0, stream>>>(w1,     wm1,   589824);
  cvt_f16<<<(589824 + 255) / 256, 256, 0, stream>>>(w2,     wm2,   589824);
  rpb_expand<<<(NHEADS * SQ * SK + 255) / 256, 256, 0, stream>>>(rpb, rpbm);
  mask_expand<<<(512 * SQ * SK + 255) / 256, 256, 0, stream>>>(mask, maskp);

  // 2) LN1 + cyclic shift + window partition (gather), f16
  ln_kernel<<<NTOK / 8, 256, 0, stream>>>(hidden, n1w, n1b, xw, 0);

  // 3) QKV GEMM (M=100352, N=1152, K=384), fused bias + q-scale + head split
  EpiQKV eq{qkv_b, qbuf, kbuf, vbuf, 0.17677669529663687f};
  gemm128x64<EpiQKV><<<dim3(18, 784), 256, 0, stream>>>(xw, wqkv, DIMC, eq);

  // 4) Windowed attention (one block per window-head)
  hipFuncSetAttribute((const void*)attn_kernel,
                      hipFuncAttributeMaxDynamicSharedMemorySize, ATTN_SMEM);
  attn_kernel<<<NWIN * NHEADS, 256, ATTN_SMEM, stream>>>(qbuf, kbuf, vbuf, rpbm, maskp, aout);

  // 5) proj GEMM + window-reverse + roll + residual -> d_out = x (f32)
  EpiProj ep{proj_b, hidden, out};
  gemm128x64<EpiProj><<<dim3(6, 784), 256, 0, stream>>>(aout, wproj, DIMC, ep);

  // 6) LN2 (token order)
  ln_kernel<<<NTOK / 8, 256, 0, stream>>>(out, n2w, n2b, xh, 1);

  // 7) MLP fc1 + GELU (M=100352, N=1536, K=384)
  EpiGelu eg{b1, hbuf};
  gemm128x64<EpiGelu><<<dim3(24, 784), 256, 0, stream>>>(xh, wm1, DIMC, eg);

  // 8) MLP fc2 + residual accumulate (M=100352, N=384, K=1536)
  EpiMlp2 e2{b2, out};
  gemm128x64<EpiMlp2><<<dim3(6, 784), 256, 0, stream>>>(hbuf, wm2, 1536, e2);
}
